// NanoPrimeBlock_56530359550731
// MI455X (gfx1250) — 16-bit and 8-bit matrix operand order repaired and checked by reading, compiled, not run on hardware
//
#include <hip/hip_runtime.h>
#include <hip/hip_bf16.h>
#include <stdint.h>

typedef _Float16 f16;
typedef __attribute__((ext_vector_type(16))) _Float16 v16h;
typedef __attribute__((ext_vector_type(8)))  _Float16 v8h;
typedef __attribute__((ext_vector_type(8)))  float    v8f;
typedef __attribute__((ext_vector_type(8)))  int      v8i;
typedef __attribute__((ext_vector_type(4)))  int      v4i;

#define BB 2
#define SS 2048
#define DD 1024
#define HH 16
#define HDIM 64
#define MM (BB*SS)
#define FF 4096
#define LNEPS 1e-5f

__device__ __forceinline__ uint32_t lds_off_of(const void* p) {
  return (uint32_t)(uintptr_t)(const __attribute__((address_space(3))) void*)p;
}
__device__ __forceinline__ void async_copy_b128(const void* gsrc, uint32_t ldsdst) {
  asm volatile("global_load_async_to_lds_b128 %0, %1, off"
               :: "v"(ldsdst), "v"(gsrc)
               : "memory");
}
__device__ __forceinline__ void wait_asynccnt0() {
  asm volatile("s_wait_asynccnt 0" ::: "memory");
}

__device__ __forceinline__ v16h load_a16(const f16* base, int ld, int row_base,
                                         int k, int lane) {
  int r  = row_base + (lane & 15);
  int hf = lane >> 4;
  const f16* p = base + (size_t)r * ld + k + hf * 8;
  union { v16h v; v8h h2[2]; } u;
  u.h2[0] = *(const v8h*)(p);
  u.h2[1] = *(const v8h*)(p + 16);
  return u.v;
}

__device__ __forceinline__ v16h load_b16(const f16* bt, int ld, int n_base,
                                         int k, int lane) {
  int n  = n_base + (lane & 15);
  int hf = lane >> 4;
  const f16* p = bt + (size_t)n * ld + k + hf * 8;
  union { v16h v; v8h h2[2]; } u;
  u.h2[0] = *(const v8h*)(p);
  u.h2[1] = *(const v8h*)(p + 16);
  return u.v;
}

__device__ __forceinline__ v8i load_a8(const int8_t* base, int ld, int row_base,
                                       int k, int lane) {
  int r  = row_base + (lane & 15);
  int hf = lane >> 4;
  const int8_t* p = base + (size_t)r * ld + k + hf * 8;
  union { v8i v; int64_t q[4]; } u;
  u.q[0] = *(const int64_t*)(p);
  u.q[1] = *(const int64_t*)(p + 16);
  u.q[2] = *(const int64_t*)(p + 32);
  u.q[3] = *(const int64_t*)(p + 48);
  return u.v;
}

__device__ __forceinline__ v8i load_b8(const int8_t* bt, int ld, int n_base,
                                       int k, int lane) {
  int n  = n_base + (lane & 15);
  int hf = lane >> 4;
  const int8_t* p = bt + (size_t)n * ld + k + hf * 8;
  union { v8i v; int64_t q[4]; } u;
  u.q[0] = *(const int64_t*)(p);
  u.q[1] = *(const int64_t*)(p + 16);
  u.q[2] = *(const int64_t*)(p + 32);
  u.q[3] = *(const int64_t*)(p + 48);
  return u.v;
}

__device__ __forceinline__ v8f wmma16(v16h a, v16h b, v8f c) {
  return __builtin_amdgcn_wmma_f32_16x16x32_f16(false, a, false, b, (short)0, c,
                                                false, false);
}
__device__ __forceinline__ v8i wmma8(v8i a, v8i b, v8i c) {
  return __builtin_amdgcn_wmma_i32_16x16x64_iu8(true, a, true, b, c, false, false);
}

__device__ __forceinline__ float gelu_tanh(float x) {
  float t = tanhf(0.7978845608028654f * (x + 0.044715f * x * x * x));
  return 0.5f * x * (1.0f + t);
}

__device__ __forceinline__ float wave_sum(float v) {
#pragma unroll
  for (int off = 16; off > 0; off >>= 1) v += __shfl_xor(v, off, 32);
  return v;
}
__device__ __forceinline__ float wave_max(float v) {
#pragma unroll
  for (int off = 16; off > 0; off >>= 1) v = fmaxf(v, __shfl_xor(v, off, 32));
  return v;
}
__device__ __forceinline__ float block_sum(float v, float* red) {
  v = wave_sum(v);
  if ((threadIdx.x & 31) == 0) red[threadIdx.x >> 5] = v;
  __syncthreads();
  float t = 0.f;
#pragma unroll
  for (int i = 0; i < 8; ++i) t += red[i];
  __syncthreads();
  return t;
}
__device__ __forceinline__ float block_max(float v, float* red) {
  v = wave_max(v);
  if ((threadIdx.x & 31) == 0) red[threadIdx.x >> 5] = v;
  __syncthreads();
  float t = red[0];
#pragma unroll
  for (int i = 1; i < 8; ++i) t = fmaxf(t, red[i]);
  __syncthreads();
  return t;
}

template <int MODE>
__global__ __launch_bounds__(256) void gemm_f16_kernel(
    const f16* __restrict__ A, const f16* __restrict__ Bt,
    f16* __restrict__ out16, float* __restrict__ out32,
    const float* __restrict__ resid, int N, int K) {
  __shared__ __align__(16) f16 btile[64 * 32];
  int lane = threadIdx.x & 31;
  int wave = threadIdx.x >> 5;
  int m_base = blockIdx.x * 128 + wave * 16;
  int n_base = blockIdx.y * 64;
  int bn = threadIdx.x >> 2;
  int bc = threadIdx.x & 3;
  uint32_t ldst = lds_off_of(&btile[0]) + (uint32_t)(bn * 64 + bc * 16);
  const f16* gsrcb = Bt + (size_t)(n_base + bn) * K + bc * 8;
  v8f acc[4] = {};
  for (int k = 0; k < K; k += 32) {
    __syncthreads();
    async_copy_b128(gsrcb + k, ldst);
    v16h a = load_a16(A, K, m_base, k, lane);
    if (k + 32 < K)
      __builtin_prefetch(A + (size_t)(m_base + (lane & 15)) * K + k + 32, 0, 0);
    wait_asynccnt0();
    __syncthreads();
#pragma unroll
    for (int t = 0; t < 4; ++t) {
      v16h b = load_b16(btile, 32, 16 * t, 0, lane);
      acc[t] = wmma16(a, b, acc[t]);
    }
  }
  int hf = lane >> 4, col = lane & 15;
#pragma unroll
  for (int t = 0; t < 4; ++t) {
    int n = n_base + 16 * t + col;
    if (MODE == 1) {
      int hh = n >> 6, hd = n & (HDIM - 1);
      int m0 = m_base + 8 * hf;
      int bbx = m0 >> 11, s0 = m0 & (SS - 1);
      __align__(16) f16 tmp[8];
#pragma unroll
      for (int r = 0; r < 8; ++r) tmp[r] = (f16)acc[t][r];
      *(v8h*)(out16 + ((size_t)((bbx * HH + hh) * HDIM + hd)) * SS + s0) =
          *(v8h*)tmp;
    } else {
#pragma unroll
      for (int r = 0; r < 8; ++r) {
        int m = m_base + r + 8 * hf;
        size_t idx = (size_t)m * N + n;
        float v = acc[t][r];
        if (MODE == 0) out16[idx] = (f16)v;
        if (MODE == 2) out32[idx] = v + resid[idx];
      }
    }
  }
}

template <int MODE>
__global__ __launch_bounds__(256) void gemm_i8_kernel(
    const int8_t* __restrict__ A, const int8_t* __restrict__ Bt,
    f16* __restrict__ out16, float* __restrict__ out32,
    const float* __restrict__ scale_x, const float* __restrict__ wsb, int wsidx,
    const float* __restrict__ resid, int N, int K) {
  __shared__ __align__(16) int8_t btile8[64 * 64];
  int lane = threadIdx.x & 31;
  int wave = threadIdx.x >> 5;
  int m_base = blockIdx.x * 128 + wave * 16;
  int n_base = blockIdx.y * 64;
  int bn = threadIdx.x >> 2;
  int bc = threadIdx.x & 3;
  uint32_t ldst = lds_off_of(&btile8[0]) + (uint32_t)(bn * 64 + bc * 16);
  const int8_t* gsrcb = Bt + (size_t)(n_base + bn) * K + bc * 16;
  float inv_ws = wsb[wsidx];
  v8i acc[4] = {};
  for (int k = 0; k < K; k += 64) {
    __syncthreads();
    async_copy_b128(gsrcb + k, ldst);
    v8i a = load_a8(A, K, m_base, k, lane);
    if (k + 64 < K)
      __builtin_prefetch(A + (size_t)(m_base + (lane & 15)) * K + k + 64, 0, 0);
    wait_asynccnt0();
    __syncthreads();
#pragma unroll
    for (int t = 0; t < 4; ++t) {
      v8i b = load_b8(btile8, 64, 16 * t, 0, lane);
      acc[t] = wmma8(a, b, acc[t]);
    }
  }
  int hf = lane >> 4, col = lane & 15;
#pragma unroll
  for (int t = 0; t < 4; ++t) {
    int n = n_base + 16 * t + col;
#pragma unroll
    for (int r = 0; r < 8; ++r) {
      int m = m_base + r + 8 * hf;
      size_t idx = (size_t)m * N + n;
      float y = (float)acc[t][r] * scale_x[m] * inv_ws;
      if (MODE == 0) out16[idx] = (f16)gelu_tanh(y);
      if (MODE == 1) out32[idx] = resid[idx] + y;
    }
  }
}

__global__ __launch_bounds__(256) void attn_kernel(
    const f16* __restrict__ Q, const f16* __restrict__ Km,
    const f16* __restrict__ Vt, f16* __restrict__ O) {
  __shared__ __align__(16) _Float16 pbuf[8][16][32];
  int lane = threadIdx.x & 31;
  int wave = threadIdx.x >> 5;
  int hf = lane >> 4, l15 = lane & 15;
  int b = blockIdx.y >> 4, h = blockIdx.y & 15;
  int q0 = blockIdx.x * 128 + wave * 16;

  const f16* Qb = Q + (size_t)b * SS * DD + h * HDIM;
  const f16* Kb = Km + (size_t)b * SS * DD + h * HDIM;
  const f16* Vb = Vt + (size_t)(b * HH + h) * HDIM * SS;

  v16h qa0 = load_a16(Qb, DD, q0, 0, lane);
  v16h qa1 = load_a16(Qb, DD, q0, 32, lane);

  v8f o[4] = {};
  float mrow[8], lrow[8];
#pragma unroll
  for (int r = 0; r < 8; ++r) { mrow[r] = -3.0e38f; lrow[r] = 0.f; }

  int kend = q0 + 16;
  for (int kc = 0; kc < kend; kc += 32) {
    v8f s0 = {}, s1 = {};
    {
      v16h b0 = load_b16(Kb, DD, kc, 0, lane);
      v16h b1 = load_b16(Kb, DD, kc, 32, lane);
      s0 = wmma16(qa0, b0, s0);
      s0 = wmma16(qa1, b1, s0);
    }
    {
      v16h b0 = load_b16(Kb, DD, kc + 16, 0, lane);
      v16h b1 = load_b16(Kb, DD, kc + 16, 32, lane);
      s1 = wmma16(qa0, b0, s1);
      s1 = wmma16(qa1, b1, s1);
    }
#pragma unroll
    for (int r = 0; r < 8; ++r) {
      int qrow = q0 + r + 8 * hf;
      float x0 = s0[r] * 0.125f;
      float x1 = s1[r] * 0.125f;
      x0 = (kc + l15 <= qrow) ? x0 : -3.0e38f;
      x1 = (kc + 16 + l15 <= qrow) ? x1 : -3.0e38f;
      float cm = fmaxf(x0, x1);
#pragma unroll
      for (int off = 1; off < 16; off <<= 1)
        cm = fmaxf(cm, __shfl_xor(cm, off, 32));
      float mnew = fmaxf(mrow[r], cm);
      float scl = __expf(mrow[r] - mnew);
      mrow[r] = mnew;
      float p0 = __expf(x0 - mnew);
      float p1 = __expf(x1 - mnew);
      float ps = p0 + p1;
#pragma unroll
      for (int off = 1; off < 16; off <<= 1) ps += __shfl_xor(ps, off, 32);
      lrow[r] = lrow[r] * scl + ps;
#pragma unroll
      for (int t = 0; t < 4; ++t) o[t][r] = o[t][r] * scl;
      pbuf[wave][r + 8 * hf][l15] = (f16)p0;
      pbuf[wave][r + 8 * hf][16 + l15] = (f16)p1;
    }
    v16h pf = load_a16((const f16*)&pbuf[wave][0][0], 32, 0, 0, lane);
#pragma unroll
    for (int t = 0; t < 4; ++t) {
      v16h vf = load_b16(Vb, SS, 16 * t, kc, lane);
      o[t] = wmma16(pf, vf, o[t]);
    }
  }
#pragma unroll
  for (int t = 0; t < 4; ++t) {
    int n = h * HDIM + 16 * t + l15;
#pragma unroll
    for (int r = 0; r < 8; ++r) {
      int m = q0 + r + 8 * hf;
      float ov = o[t][r] / lrow[r];
      O[((size_t)(b * SS + m)) * DD + n] = (f16)ov;
    }
  }
}

__global__ __launch_bounds__(256) void ln_f16_kernel(
    const float* __restrict__ X, const float* __restrict__ w,
    const float* __restrict__ bb, f16* __restrict__ out) {
  __shared__ float red[8];
  int row = blockIdx.x;
  const float* x = X + (size_t)row * DD;
  float v[4], s = 0.f, s2 = 0.f;
#pragma unroll
  for (int i = 0; i < 4; ++i) {
    v[i] = x[threadIdx.x + 256 * i];
    s += v[i];
    s2 += v[i] * v[i];
  }
  s = block_sum(s, red);
  s2 = block_sum(s2, red);
  float mu = s * (1.f / DD);
  float inv = rsqrtf(s2 * (1.f / DD) - mu * mu + LNEPS);
#pragma unroll
  for (int i = 0; i < 4; ++i) {
    int c = threadIdx.x + 256 * i;
    out[(size_t)row * DD + c] = (f16)((v[i] - mu) * inv * w[c] + bb[c]);
  }
}

__global__ __launch_bounds__(256) void ln_quant_kernel(
    const float* __restrict__ X, const float* __restrict__ w,
    const float* __restrict__ bb, int8_t* __restrict__ out,
    float* __restrict__ scale_x) {
  __shared__ float red[8];
  int row = blockIdx.x;
  const float* x = X + (size_t)row * DD;
  float v[4], s = 0.f, s2 = 0.f;
#pragma unroll
  for (int i = 0; i < 4; ++i) {
    v[i] = x[threadIdx.x + 256 * i];
    s += v[i];
    s2 += v[i] * v[i];
  }
  s = block_sum(s, red);
  s2 = block_sum(s2, red);
  float mu = s * (1.f / DD);
  float inv = rsqrtf(s2 * (1.f / DD) - mu * mu + LNEPS);
  float hv[4], amax = 0.f;
#pragma unroll
  for (int i = 0; i < 4; ++i) {
    int c = threadIdx.x + 256 * i;
    hv[i] = (v[i] - mu) * inv * w[c] + bb[c];
    amax = fmaxf(amax, fabsf(hv[i]));
  }
  amax = block_max(amax, red);
  float mx = fmaxf(amax, LNEPS);
  float xs = 127.f / mx;
#pragma unroll
  for (int i = 0; i < 4; ++i) {
    int c = threadIdx.x + 256 * i;
    float q = fminf(127.f, fmaxf(-128.f, rintf(hv[i] * xs)));
    out[(size_t)row * DD + c] = (int8_t)q;
  }
  if (threadIdx.x == 0) scale_x[row] = mx / 127.f;
}

__global__ __launch_bounds__(256) void quant_act_kernel(
    const f16* __restrict__ G, int8_t* __restrict__ out,
    float* __restrict__ scale_x) {
  __shared__ float red[8];
  int row = blockIdx.x;
  const f16* g = G + (size_t)row * FF;
  float hv[16], amax = 0.f;
#pragma unroll
  for (int i = 0; i < 16; ++i) {
    hv[i] = (float)g[threadIdx.x + 256 * i];
    amax = fmaxf(amax, fabsf(hv[i]));
  }
  amax = block_max(amax, red);
  float mx = fmaxf(amax, LNEPS);
  float xs = 127.f / mx;
#pragma unroll
  for (int i = 0; i < 16; ++i) {
    int c = threadIdx.x + 256 * i;
    float q = fminf(127.f, fmaxf(-128.f, rintf(hv[i] * xs)));
    out[(size_t)row * FF + c] = (int8_t)q;
  }
  if (threadIdx.x == 0) scale_x[row] = mx / 127.f;
}

__global__ __launch_bounds__(256) void cast_t_kernel(const float* __restrict__ W,
                                                     f16* __restrict__ Wt, int K,
                                                     int N) {
  int idx = blockIdx.x * 256 + threadIdx.x;
  if (idx >= K * N) return;
  int n = idx / K, k = idx - n * K;
  Wt[idx] = (f16)W[(size_t)k * N + n];
}

__global__ __launch_bounds__(256) void quant_w_kernel(
    const float* __restrict__ W, int8_t* __restrict__ Wt,
    const float* __restrict__ wsb, int wsidx, int K, int N) {
  int idx = blockIdx.x * 256 + threadIdx.x;
  if (idx >= K * N) return;
  float wsc = 1.f / wsb[wsidx];
  int n = idx / K, k = idx - n * K;
  float q = rintf(W[(size_t)k * N + n] * wsc);
  Wt[idx] = (int8_t)fminf(1.f, fmaxf(-1.f, q));
}

__global__ __launch_bounds__(256) void absmean_partial_kernel(
    const float* __restrict__ W, float* __restrict__ part, int n) {
  __shared__ float red[8];
  int base = blockIdx.x * 4096;
  float s = 0.f;
  for (int i = threadIdx.x; i < 4096; i += 256) {
    int idx = base + i;
    if (idx < n) s += fabsf(W[idx]);
  }
  s = block_sum(s, red);
  if (threadIdx.x == 0) part[blockIdx.x] = s;
}

__global__ __launch_bounds__(256) void absmean_final_kernel(
    const float* __restrict__ p1, const float* __restrict__ p2,
    float* __restrict__ wsb, float inv_n1, float inv_n2) {
  __shared__ float red[8];
  float s1 = 0.f, s2 = 0.f;
  for (int i = threadIdx.x; i < 1024; i += 256) {
    s1 += p1[i];
    s2 += p2[i];
  }
  s1 = block_sum(s1, red);
  s2 = block_sum(s2, red);
  if (threadIdx.x == 0) {
    wsb[0] = fmaxf(s1 * inv_n1, LNEPS);
    wsb[1] = fmaxf(s2 * inv_n2, LNEPS);
  }
}

extern "C" void kernel_launch(void* const* d_in, const int* in_sizes, int n_in,
                              void* d_out, int out_size, void* d_ws,
                              size_t ws_size, hipStream_t stream) {
  (void)in_sizes; (void)n_in; (void)out_size; (void)ws_size;
  const float* x    = (const float*)d_in[0];
  const float* n1w  = (const float*)d_in[1];
  const float* n1b  = (const float*)d_in[2];
  const float* n2w  = (const float*)d_in[3];
  const float* n2b  = (const float*)d_in[4];
  const float* wqd  = (const float*)d_in[5];
  const float* wqu  = (const float*)d_in[6];
  const float* wkvd = (const float*)d_in[7];
  const float* wku  = (const float*)d_in[8];
  const float* wvu  = (const float*)d_in[9];
  const float* wo   = (const float*)d_in[10];
  const float* w1   = (const float*)d_in[11];
  const float* w2   = (const float*)d_in[12];
  float* out = (float*)d_out;

  char* ws = (char*)d_ws;
  const size_t MB = 1ull << 20;
  f16* h1     = (f16*)(ws + 0);
  f16* wqd_t  = (f16*)(ws + 8 * MB);
  f16* wqu_t  = (f16*)(ws + 9 * MB);
  f16* wkvd_t = (f16*)(ws + 10 * MB);
  f16* wku_t  = (f16*)(ws + 10 * MB + 512 * 1024);
  f16* wvu_t  = (f16*)(ws + 11 * MB);
  f16* wo_t   = (f16*)(ws + 11 * MB + 512 * 1024);
  f16* qlat   = (f16*)(ws + 14 * MB);
  f16* qbuf   = (f16*)(ws + 18 * MB);
  f16* ckv    = (f16*)(ws + 26 * MB);
  f16* kbuf   = (f16*)(ws + 28 * MB);
  f16* vt     = (f16*)(ws + 36 * MB);
  f16* attn   = (f16*)(ws + 44 * MB);
  int8_t* w1q = (int8_t*)(ws + 0);
  int8_t* w2q = (int8_t*)(ws + 4 * MB);
  int8_t* h2q = (int8_t*)(ws + 8 * MB);
  float* scale1 = (float*)(ws + 12 * MB);
  float* scale2 = (float*)(ws + 12 * MB + 64 * 1024);
  float* wsb    = (float*)(ws + 12 * MB + 128 * 1024);
  float* part1  = (float*)(ws + 12 * MB + 192 * 1024);
  float* part2  = (float*)(ws + 12 * MB + 256 * 1024);
  f16* gbuf   = (f16*)(ws + 13 * MB);
  int8_t* gq  = (int8_t*)(ws + 45 * MB);
  float* x1   = (float*)(ws + 61 * MB);

  dim3 blk(256);
  ln_f16_kernel<<<MM, blk, 0, stream>>>(x, n1w, n1b, h1);
  cast_t_kernel<<<(DD * 512 + 255) / 256, blk, 0, stream>>>(wqd, wqd_t, DD, 512);
  cast_t_kernel<<<(512 * DD + 255) / 256, blk, 0, stream>>>(wqu, wqu_t, 512, DD);
  cast_t_kernel<<<(DD * 256 + 255) / 256, blk, 0, stream>>>(wkvd, wkvd_t, DD, 256);
  cast_t_kernel<<<(256 * DD + 255) / 256, blk, 0, stream>>>(wku, wku_t, 256, DD);
  cast_t_kernel<<<(256 * DD + 255) / 256, blk, 0, stream>>>(wvu, wvu_t, 256, DD);
  cast_t_kernel<<<(DD * DD + 255) / 256, blk, 0, stream>>>(wo, wo_t, DD, DD);
  gemm_f16_kernel<0><<<dim3(MM / 128, 512 / 64), blk, 0, stream>>>(
      h1, wqd_t, qlat, nullptr, nullptr, 512, DD);
  gemm_f16_kernel<0><<<dim3(MM / 128, DD / 64), blk, 0, stream>>>(
      qlat, wqu_t, qbuf, nullptr, nullptr, DD, 512);
  gemm_f16_kernel<0><<<dim3(MM / 128, 256 / 64), blk, 0, stream>>>(
      h1, wkvd_t, ckv, nullptr, nullptr, 256, DD);
  gemm_f16_kernel<0><<<dim3(MM / 128, DD / 64), blk, 0, stream>>>(
      ckv, wku_t, kbuf, nullptr, nullptr, DD, 256);
  gemm_f16_kernel<1><<<dim3(MM / 128, DD / 64), blk, 0, stream>>>(
      ckv, wvu_t, vt, nullptr, nullptr, DD, 256);
  attn_kernel<<<dim3(SS / 128, BB * HH), blk, 0, stream>>>(qbuf, kbuf, vt, attn);
  gemm_f16_kernel<2><<<dim3(MM / 128, DD / 64), blk, 0, stream>>>(
      attn, wo_t, nullptr, x1, x, DD, DD);
  absmean_partial_kernel<<<1024, blk, 0, stream>>>(w1, part1, DD * FF);
  absmean_partial_kernel<<<1024, blk, 0, stream>>>(w2, part2, FF * DD);
  absmean_final_kernel<<<1, blk, 0, stream>>>(part1, part2, wsb,
                                              1.0f / (DD * FF), 1.0f / (FF * DD));
  quant_w_kernel<<<(DD * FF + 255) / 256, blk, 0, stream>>>(w1, w1q, wsb, 0, DD, FF);
  quant_w_kernel<<<(FF * DD + 255) / 256, blk, 0, stream>>>(w2, w2q, wsb, 1, FF, DD);
  ln_quant_kernel<<<MM, blk, 0, stream>>>(x1, n2w, n2b, h2q, scale1);
  gemm_i8_kernel<0><<<dim3(MM / 128, FF / 64), blk, 0, stream>>>(
      h2q, w1q, gbuf, nullptr, scale1, wsb, 0, nullptr, FF, DD);
  quant_act_kernel<<<MM, blk, 0, stream>>>(gbuf, gq, scale2);
  gemm_i8_kernel<1><<<dim3(MM / 128, DD / 64), blk, 0, stream>>>(
      gq, w2q, nullptr, out, scale2, wsb, 1, x1, DD, FF);
}
